// SolutionCalculatorVolume_49435073577132
// MI455X (gfx1250) — matrix operand order repaired and checked by reading, not compiled, not run on hardware
//
#include <hip/hip_runtime.h>
#include <hip/hip_bf16.h>
#include <math.h>

// ---------------------------------------------------------------------------
// MI455X (gfx1250): f16 WMMA (f32 accumulate) for all 256-K GEMMs.
// - 4 independent accumulator chains per wave to hide the f16 WMMA->WMMA
//   hazard (ISA 7.12.1).
// - Packed weights laid out so that all 32 fragment loads of a K-loop come
//   from ONE base pointer + constant 24-bit instruction offsets (no per-load
//   64-bit address math -> no VGPR spills).
// ---------------------------------------------------------------------------

typedef __attribute__((ext_vector_type(16))) _Float16 v16h;
typedef __attribute__((ext_vector_type(8)))  _Float16 v8h;
typedef __attribute__((ext_vector_type(8)))  float    v8f;

#define NPTS   8192
#define SS     9        // 1 center + 8 neighbors
#define VV     5
#define HD     256
#define KSAMP  8
#define RNOISE (1.0f / 50.0f)
#define BT     128      // 4 waves per block for the GEMM kernels

// Packed B layout strides (in _Float16 elements):
//   element  = lane*16 + e            (32B per lane, contiguous)
//   nt tile  stride = 32*16   = 512   (1 KB)
//   kk step  stride = 16*512  = 8192  (16 KB)
//   v        stride = 8*8192  = 65536
#define NT_STRIDE 512
#define KK_STRIDE 8192

__device__ __forceinline__ uint32_t pcg_hash(uint32_t x) {
  x = x * 747796405u + 2891336453u;
  uint32_t w = ((x >> ((x >> 28) + 4u)) ^ x) * 277803737u;
  return (w >> 22) ^ w;
}
__device__ __forceinline__ float u01(uint32_t h) {
  return (float)(h >> 8) * (1.0f / 16777216.0f);
}

// A-fragment (16x32 f16, ISA layout) from a per-lane LDS base:
// base = lds + (lane%16)*HD + ((lane&16)?8:0); kk adds 32 elements.
__device__ __forceinline__ v16h load_afrag(const _Float16* lane_base, int kk) {
  v8h lo = *(const v8h*)(lane_base + kk * 32);
  v8h hi = *(const v8h*)(lane_base + kk * 32 + 16);
  v16h a;
#pragma unroll
  for (int i = 0; i < 8; ++i) { a[i] = lo[i]; a[i + 8] = hi[i]; }
  return a;
}

#define WMMA_F16(A, B, C) \
  __builtin_amdgcn_wmma_f32_16x16x32_f16(false, (A), false, (B), (short)0, (C), false, false)

// One K=256 GEMM macro body: 8 kk steps, 4 N-tiles, base-pointer addressing.
#define GEMM_K256(ABASE, BBASE, C0, C1, C2, C3)                           \
  _Pragma("unroll")                                                        \
  for (int kk = 0; kk < 8; ++kk) {                                         \
    v16h a  = load_afrag((ABASE), kk);                                     \
    v16h b0 = *(const v16h*)((BBASE) + kk * KK_STRIDE + 0 * NT_STRIDE);    \
    v16h b1 = *(const v16h*)((BBASE) + kk * KK_STRIDE + 1 * NT_STRIDE);    \
    v16h b2 = *(const v16h*)((BBASE) + kk * KK_STRIDE + 2 * NT_STRIDE);    \
    v16h b3 = *(const v16h*)((BBASE) + kk * KK_STRIDE + 3 * NT_STRIDE);    \
    C0 = WMMA_F16(a, b0, C0);                                              \
    C1 = WMMA_F16(a, b1, C1);                                              \
    C2 = WMMA_F16(a, b2, C2);                                              \
    C3 = WMMA_F16(a, b3, C3);                                              \
  }

// ---------------------------------------------------------------------------
__global__ void zero_out_kernel(float* out, int n) {
  int i = blockIdx.x * blockDim.x + threadIdx.x;
  if (i < n) out[i] = 0.0f;
}

// Pack a (V,256,256) f32 [v][k][n] weight into f16 WMMA B-fragment order.
// B-matrix 32x16 layout: lane<16 -> K=kk*32+{0..7,16..23}, lane>=16 -> K=kk*32+{8..15,24..31};
// 2 values per VGPR; N = nt*16 + lane%16.
__global__ void pack_weights_kernel(const float* Wb2, const float* Wa1b,
                                    const float* Wa1n, const float* Wa1g,
                                    _Float16* dst) {
  const int t = blockIdx.x * blockDim.x + threadIdx.x;   // 0..327679
  const int which = blockIdx.y;
  const float* src = (which == 0) ? Wb2 : (which == 1) ? Wa1b
                   : (which == 2) ? Wa1n : Wa1g;
  const int e    = t & 15;
  const int lane = (t >> 4) & 31;
  const int nt   = (t >> 9) & 15;
  const int kk   = (t >> 13) & 7;
  const int v    = t >> 16;
  if (v >= VV) return;
  const int r = e >> 1, h2 = e & 1;
  const int k = kk * 32 + ((lane & 16) ? 8 : 0) + ((e & 8) ? 16 : 0) + ((2 * r + h2) & 7);
  const int n = nt * 16 + (lane & 15);
  dst[which * (VV * HD * HD) + t] = (_Float16)src[(v * HD + k) * HD + n];
}

// pe = relu(ratio @ Wp + bp); peW[v][h] = pe @ Wa1_param[v] + ba1[v]
__global__ void pe_enc_base_kernel(const float* gpv, const float* gpr,
                                   const float* Wp, const float* bp,
                                   const float* Wa1_param, const float* ba1,
                                   float* peW) {
  __shared__ float pe[HD];
  const int t = threadIdx.x;
  const float r0 = gpv[0] / gpr[0];
  const float r1 = gpv[1] / gpr[1];
  pe[t] = fmaxf(r0 * Wp[t] + r1 * Wp[HD + t] + bp[t], 0.0f);
  __syncthreads();
  for (int v = 0; v < VV; ++v) {
    float acc = ba1[v * HD + t];
    for (int d = 0; d < HD; ++d)
      acc += pe[d] * Wa1_param[(v * HD + d) * HD + t];
    peW[v * HD + t] = acc;
  }
}

// enc[n][v][h] = node @ Wa1_node[v] + geo @ Wa1_geo[v] + peW[v]
__global__ void enc_gemm_kernel(const float* enc_node, const float* enc_geo,
                                const _Float16* pkN, const _Float16* pkG,
                                const float* peW, float* enc) {
  __shared__ alignas(16) _Float16 An[16 * HD];
  __shared__ alignas(16) _Float16 Ag[16 * HD];
  const int tile = blockIdx.x, v = blockIdx.y;
  const int n0 = tile * 16;
  const int tid = threadIdx.x;
  for (int e = tid; e < 16 * HD; e += BT) {
    An[e] = (_Float16)enc_node[n0 * HD + e];
    Ag[e] = (_Float16)enc_geo[n0 * HD + e];
  }
  __syncthreads();

  const int w = tid >> 5, lane = tid & 31;
  const int nt0 = w * 4;
  const int alane = (lane & 15) * HD + ((lane & 16) ? 8 : 0);
  const _Float16* bbN = pkN + (size_t)v * (VV == 0 ? 0 : 8 * KK_STRIDE)
                            + nt0 * NT_STRIDE + lane * 16;
  const _Float16* bbG = pkG + (size_t)v * 8 * KK_STRIDE
                            + nt0 * NT_STRIDE + lane * 16;

  v8f c0 = {}, c1 = {}, c2 = {}, c3 = {};
  GEMM_K256(An + alane, bbN, c0, c1, c2, c3);
  GEMM_K256(Ag + alane, bbG, c0, c1, c2, c3);

  // C/D layout: VGPR r -> M = r (+8 for lanes>=16), N = lane%16 (+nt*16)
  const int cb = lane & 15;
#pragma unroll
  for (int r = 0; r < 8; ++r) {
    const int m = (lane & 16) ? r + 8 : r;
    const size_t eb = (size_t)(n0 + m) * (VV * HD) + (size_t)v * HD;
    const int cc = nt0 * 16 + cb;
    enc[eb + cc +  0] = c0[r] + peW[v * HD + cc +  0];
    enc[eb + cc + 16] = c1[r] + peW[v * HD + cc + 16];
    enc[eb + cc + 32] = c2[r] + peW[v * HD + cc + 32];
    enc[eb + cc + 48] = c3[r] + peW[v * HD + cc + 48];
  }
}

// Fused per-(n,s) pipeline: coords -> h1 (VALU, K=3) -> h2 (WMMA) ->
// a1 (WMMA + enc) -> dot(Wa2) -> weighted atomic accumulate into out[n][v].
__global__ void fuse_main_kernel(const float* centers,
                                 const _Float16* pkWb2, const _Float16* pkWa1b,
                                 const float* Wb1, const float* bb1,
                                 const float* bb2, const float* enc,
                                 const float* Wa2, const float* ba2,
                                 float* out) {
  __shared__ float coords[16][3];
  __shared__ int   rowN[16];               // row -> point index n
  __shared__ alignas(16) _Float16 bufA[16 * HD];
  __shared__ alignas(16) _Float16 bufB[16 * HD];
  __shared__ float red[16][8];

  const int tile = blockIdx.x, v = blockIdx.y;
  const int tid = threadIdx.x;

  // --- sphere sampling (deterministic hash RNG) + row->n table ---
  if (tid < 16) {
    const int g = tile * 16 + tid;
    const int n = g / SS, s = g - n * SS;
    rowN[tid] = n;
    float cx = centers[n * 3 + 0], cy = centers[n * 3 + 1], cz = centers[n * 3 + 2];
    if (s > 0) {
      const uint32_t q = (uint32_t)(n * KSAMP + (s - 1)) * 6u;
      float dx = u01(pcg_hash(q + 0)) - 0.5f;
      float dy = u01(pcg_hash(q + 1)) - 0.5f;
      float dz = u01(pcg_hash(q + 2)) - 0.5f;
      const float inv = rsqrtf(dx * dx + dy * dy + dz * dz + 1e-12f);
      const float r0 = RNOISE * cbrtf(u01(pcg_hash(q + 3)));
      const float r1 = RNOISE * cbrtf(u01(pcg_hash(q + 4)));
      const float r2 = RNOISE * cbrtf(u01(pcg_hash(q + 5)));
      cx += dx * inv * r0; cy += dy * inv * r1; cz += dz * inv * r2;
    }
    coords[tid][0] = cx; coords[tid][1] = cy; coords[tid][2] = cz;
  }
  __syncthreads();

  // --- h1 = relu(coords @ Wb1[v] + bb1[v]), K=3 on VALU (2 cols/thread) ---
#pragma unroll
  for (int cc = 0; cc < 2; ++cc) {
    const int col = tid + cc * BT;
    const float w0 = Wb1[(v * 3 + 0) * HD + col];
    const float w1 = Wb1[(v * 3 + 1) * HD + col];
    const float w2 = Wb1[(v * 3 + 2) * HD + col];
    const float b  = bb1[v * HD + col];
#pragma unroll
    for (int m = 0; m < 16; ++m) {
      const float x = coords[m][0] * w0 + coords[m][1] * w1 + coords[m][2] * w2 + b;
      bufA[m * HD + col] = (_Float16)fmaxf(x, 0.0f);
    }
  }
  __syncthreads();

  const int w = tid >> 5, lane = tid & 31;
  const int nt0 = w * 4;
  const int cb = lane & 15;
  const int alane = (lane & 15) * HD + ((lane & 16) ? 8 : 0);
  const int boff = nt0 * NT_STRIDE + lane * 16;
  const _Float16* bb_1 = pkWb2  + (size_t)v * 8 * KK_STRIDE + boff;
  const _Float16* bb_2 = pkWa1b + (size_t)v * 8 * KK_STRIDE + boff;

  // --- GEMM1: h2 = relu(h1 @ Wb2[v] + bb2[v]) ---
  {
    v8f c0 = {}, c1 = {}, c2 = {}, c3 = {};
    GEMM_K256(bufA + alane, bb_1, c0, c1, c2, c3);
#pragma unroll
    for (int r = 0; r < 8; ++r) {
      const int m  = (lane & 16) ? r + 8 : r;
      const int cc = nt0 * 16 + cb;
      bufB[m * HD + cc +  0] = (_Float16)fmaxf(c0[r] + bb2[v * HD + cc +  0], 0.0f);
      bufB[m * HD + cc + 16] = (_Float16)fmaxf(c1[r] + bb2[v * HD + cc + 16], 0.0f);
      bufB[m * HD + cc + 32] = (_Float16)fmaxf(c2[r] + bb2[v * HD + cc + 32], 0.0f);
      bufB[m * HD + cc + 48] = (_Float16)fmaxf(c3[r] + bb2[v * HD + cc + 48], 0.0f);
    }
  }
  __syncthreads();

  // --- GEMM2: a1 = relu(h2 @ Wa1_basis[v] + enc[n][v]) ---
  {
    v8f c0 = {}, c1 = {}, c2 = {}, c3 = {};
    GEMM_K256(bufB + alane, bb_2, c0, c1, c2, c3);
#pragma unroll
    for (int r = 0; r < 8; ++r) {
      const int m = (lane & 16) ? r + 8 : r;
      const size_t eb = (size_t)rowN[m] * (VV * HD) + (size_t)v * HD;
      const int cc = nt0 * 16 + cb;
      bufA[m * HD + cc +  0] = (_Float16)fmaxf(c0[r] + enc[eb + cc +  0], 0.0f);
      bufA[m * HD + cc + 16] = (_Float16)fmaxf(c1[r] + enc[eb + cc + 16], 0.0f);
      bufA[m * HD + cc + 32] = (_Float16)fmaxf(c2[r] + enc[eb + cc + 32], 0.0f);
      bufA[m * HD + cc + 48] = (_Float16)fmaxf(c3[r] + enc[eb + cc + 48], 0.0f);
    }
  }
  __syncthreads();

  // --- out_row[m] = a1[m] . Wa2[v] + ba2[v]; weighted sample average ---
  {
    const int m = tid >> 3, ch = tid & 7;   // 16 rows x 8 chunks of 32
    float p = 0.0f;
#pragma unroll
    for (int j = 0; j < 32; ++j)
      p += (float)bufA[m * HD + ch * 32 + j] * Wa2[v * HD + ch * 32 + j];
    red[m][ch] = p;
  }
  __syncthreads();
  if (tid < 16) {
    float sum = ba2[v];
#pragma unroll
    for (int j = 0; j < 8; ++j) sum += red[tid][j];
    const int g = tile * 16 + tid;
    const int n = g / SS, s = g - n * SS;
    const float wt = (s == 0) ? 0.5f : (0.5f / 8.0f);
    atomicAdd(&out[n * VV + v], wt * sum);
  }
}

// ---------------------------------------------------------------------------
extern "C" void kernel_launch(void* const* d_in, const int* in_sizes, int n_in,
                              void* d_out, int out_size, void* d_ws, size_t ws_size,
                              hipStream_t stream) {
  const float* centers   = (const float*)d_in[0];
  const float* enc_g     = (const float*)d_in[1];
  const float* enc_node  = (const float*)d_in[2];
  const float* gpv       = (const float*)d_in[3];
  const float* gpr       = (const float*)d_in[4];
  const float* Wb1       = (const float*)d_in[5];
  const float* bb1       = (const float*)d_in[6];
  const float* Wb2       = (const float*)d_in[7];
  const float* bb2       = (const float*)d_in[8];
  const float* Wp        = (const float*)d_in[9];
  const float* bp        = (const float*)d_in[10];
  const float* Wa1_basis = (const float*)d_in[11];
  const float* Wa1_node  = (const float*)d_in[12];
  const float* Wa1_geo   = (const float*)d_in[13];
  const float* Wa1_param = (const float*)d_in[14];
  const float* ba1       = (const float*)d_in[15];
  const float* Wa2       = (const float*)d_in[16];
  const float* ba2       = (const float*)d_in[17];
  float* out = (float*)d_out;

  const int PK = VV * HD * HD;  // 327680 elements per packed tensor
  _Float16* pkWb2  = (_Float16*)d_ws;
  _Float16* pkWa1b = pkWb2 + PK;
  _Float16* pkWa1n = pkWa1b + PK;
  _Float16* pkWa1g = pkWa1n + PK;
  float* peW = (float*)(pkWa1g + PK);     // V*HD floats
  float* enc = peW + VV * HD;             // N*V*HD floats (~42 MB, L2-resident)

  zero_out_kernel<<<(NPTS * VV + 255) / 256, 256, 0, stream>>>(out, NPTS * VV);
  pack_weights_kernel<<<dim3(PK / 256, 4), 256, 0, stream>>>(
      Wb2, Wa1_basis, Wa1_node, Wa1_geo, pkWb2);
  pe_enc_base_kernel<<<1, 256, 0, stream>>>(gpv, gpr, Wp, bp, Wa1_param, ba1, peW);
  enc_gemm_kernel<<<dim3(NPTS / 16, VV), BT, 0, stream>>>(
      enc_node, enc_g, pkWa1n, pkWa1g, peW, enc);
  fuse_main_kernel<<<dim3((NPTS * SS) / 16, VV), BT, 0, stream>>>(
      centers, pkWb2, pkWa1b, Wb1, bb1, bb2, enc, Wa2, ba2, out);
}
